// ConvDicoLearningCNN_15032385536079
// MI455X (gfx1250) — matrix operand order repaired and checked by reading, not compiled, not run on hardware
//
#include <hip/hip_runtime.h>
#include <hip/hip_bf16.h>
#include <math.h>

typedef __attribute__((ext_vector_type(16))) _Float16 v16h;
typedef __attribute__((ext_vector_type(8)))  float    v8f;
typedef __attribute__((ext_vector_type(4)))  unsigned int u32x4;
typedef __attribute__((ext_vector_type(8)))  int      i32x8;
typedef __attribute__((ext_vector_type(4)))  int      i32x4;

#define PX 164
#define PY 164
#define PT 24
#define NPAD (PX*PY*PT)      /* 645504 */
#define IX 160
#define IY 160
#define IT 20
#define IMS (IX*IY*IT)       /* 512000 */
#define MB 2
#define NF 16
#define NH 8

#define YTILES (PY/2)                    /* 82 */
#define TTILES (PT/8)                    /* 3  */
#define TILES_PER_MB (PX*YTILES*TTILES)  /* 40344 */
#define TOTAL_TILES (MB*TILES_PER_MB)    /* 80688 = 8*10086 */

#if defined(__has_builtin)
#  if __has_builtin(__builtin_amdgcn_tensor_load_to_lds) && \
      __has_builtin(__builtin_amdgcn_s_wait_tensorcnt)
#    define HAVE_TDM 1
#  endif
#endif
#ifndef HAVE_TDM
#  define HAVE_TDM 0
#endif

__device__ __forceinline__ int wrapi(int v, int n) { return v < 0 ? v + n : (v >= n ? v - n : v); }
__device__ __forceinline__ float softplusf(float v) { return log1pf(__expf(v)); }

// LDS halo offset contributed by K index kk (c*128 + tau); lane adds dyp*12+dtp.
__device__ constexpr int tapOff(int kk) {
  int c = kk >> 7, tau = kk & 127;
  int tc = (tau < 125) ? tau : 0;          // W rows are zero there; value irrelevant
  int dx = tc / 25, r = tc % 25;
  int dy = r / 5,  dt = r % 5;
  return ((c * 5 + dx) * 6 + dy) * 12 + dt;
}

#if HAVE_TDM
// 1-D Tensor-DMA load of nElems f32 from global into LDS (wave-level op).
__device__ __forceinline__ void tdm_load_f32(void* ldsDst, const void* gsrc, unsigned nElems) {
  // Generic LDS pointer: low 32 bits are the LDS byte address (aperture in high bits).
  unsigned ldsAddr = (unsigned)(unsigned long long)(size_t)ldsDst;
  unsigned long long ga = (unsigned long long)(size_t)gsrc;
  u32x4 g0;
  g0[0] = 1u;                                            // count=1, user descriptor
  g0[1] = ldsAddr;                                       // lds_addr
  g0[2] = (unsigned)(ga & 0xffffffffull);                // global_addr[31:0]
  g0[3] = (unsigned)((ga >> 32) & 0x1ffffffull) | (2u << 30); // addr[56:32] | type=2
  i32x8 g1;
  g1[0] = (int)(2u << 16);                               // workgroup_mask=0, data_size=4B
  g1[1] = (int)((nElems & 0xffffu) << 16);               // tensor_dim0[15:0]
  g1[2] = (int)((nElems >> 16) & 0xffffu) | (1 << 16);   // tensor_dim0[31:16], tensor_dim1=1
  g1[3] = (int)((nElems & 0xffffu) << 16);               // tile_dim0 = nElems
  g1[4] = 1;                                             // tile_dim1 = 1
  g1[5] = (int)nElems;                                   // tensor_dim0_stride[31:0]
  g1[6] = 0;
  g1[7] = 0;
  i32x4 gz = {0, 0, 0, 0};
#if __clang_major__ >= 23
  i32x8 gz8 = {0, 0, 0, 0, 0, 0, 0, 0};
  __builtin_amdgcn_tensor_load_to_lds(g0, g1, gz, gz, gz8, 0);
#else
  __builtin_amdgcn_tensor_load_to_lds(g0, g1, gz, gz, 0);
#endif
  __builtin_amdgcn_s_wait_tensorcnt(0);
}
#endif

// ---------------------------------------------------------------------------
// 2-channel -> 16-channel circular correlation via WMMA implicit GEMM.
//   mode 0 (RHS): in = x (IM layout, zero-padded), out b_k = corr(x,d_k) + gamma*(z_k-u_k)
//   mode 1 (S)  : in = t (padded layout), out b_k = (1/g)*b_k - (1/g^2)*corr(t,d_k)  (in place)
// GEMM: C[16 filters][16 spatial] += W[16][K=256] * Patch[K=256][16]
// ---------------------------------------------------------------------------
__global__ __launch_bounds__(256) void corr16_wmma_kernel(
    const float* __restrict__ inField,
    const float* __restrict__ dHalf,     // (8,125) f32
    const float* __restrict__ zBuf,      // mode 0 only
    const float* __restrict__ uBuf,      // mode 0 only
    float* __restrict__ outB,            // (mb,16,NPAD)
    const float* __restrict__ lamP,
    const float* __restrict__ betP,
    int mode)
{
  __shared__ float    sD[NH * 125];           // raw filters (TDM target)
  __shared__ _Float16 sW[NF * 256];           // zero-padded filter matrix (f16)
  __shared__ _Float16 sIn[8][2 * 5 * 6 * 12]; // per-wave input halo, 720 f16

  const int tid  = threadIdx.x;
  const int lane = tid & 31;
  const int wv   = tid >> 5;

#if HAVE_TDM
  if (wv == 0) tdm_load_f32(sD, dHalf, NH * 125);
#else
  for (int idx = tid; idx < NH * 125; idx += 256) sD[idx] = dHalf[idx];
#endif
  __syncthreads();

  // Filter matrix: W[m][c*128+tau] = (c==m>>3 && tau<125) ? d_half[m&7][tau] : 0
  for (int idx = tid; idx < NF * 256; idx += 256) {
    int m = idx >> 8, k = idx & 255;
    int c = k >> 7, tau = k & 127;
    _Float16 v = (_Float16)0.0f;
    if (c == (m >> 3) && tau < 125) v = (_Float16)sD[(m & 7) * 125 + tau];
    sW[idx] = v;
  }

  // Tile decomposition: (mb, x, y-pair, t-octet); one 16-output tile per wave.
  int rem = blockIdx.x * 8 + wv;
  const int mbI = rem / TILES_PER_MB; rem -= mbI * TILES_PER_MB;
  const int ox  = rem / (YTILES * TTILES); rem -= ox * (YTILES * TTILES);
  const int y0  = (rem / TTILES) * 2;
  const int t0  = (rem % TTILES) * 8;

  // Stage input halo (x:5, y:6, t:12 per channel) with circular wrap applied here.
  for (int idx = lane; idx < 720; idx += 32) {
    int c = idx / 360, r = idx - c * 360;
    int dx = r / 72;  r -= dx * 72;
    int yy = r / 12;
    int tt = r - yy * 12;
    int gx = wrapi(ox - 2 + dx, PX);
    int gy = wrapi(y0 - 2 + yy, PY);
    int gt = wrapi(t0 - 2 + tt, PT);
    float v;
    if (mode == 0) {
      int ix = gx - 2, iy = gy - 2, it = gt - 2;
      v = ((unsigned)ix < IX && (unsigned)iy < IY && (unsigned)it < IT)
            ? inField[(size_t)(mbI * 2 + c) * IMS + (size_t)(ix * IY + iy) * IT + it]
            : 0.0f;
    } else {
      v = inField[(size_t)(mbI * 2 + c) * NPAD + (size_t)(gx * PY + gy) * PT + gt];
    }
    sIn[wv][idx] = (_Float16)v;
  }
  __syncthreads();

  const int p   = lane & 15;  // spatial column of this lane
  const int dyp = p >> 3;
  const int dtp = p & 7;
  const bool lo = (lane < 16);
  // A row pointer with half-wave K-offset folded in => compile-time element offsets.
  const _Float16* __restrict__ pA = sW + (lane & 15) * 256 + (lo ? 0 : 16);
  // B base pointer with lane's spatial offset folded in => constant tapOff() offsets.
  const _Float16* __restrict__ pB = sIn[wv] + dyp * 12 + dtp;

  v8f acc0 = {}, acc1 = {};
#pragma unroll
  for (int kc = 0; kc < 8; ++kc) {
    v16h a, b;
#pragma unroll
    for (int i = 0; i < 16; ++i)
      a[i] = pA[kc * 32 + i];
    if (lo) {
#pragma unroll
      for (int i = 0; i < 16; ++i) b[i] = pB[tapOff(kc * 32 + i)];
    } else {
#pragma unroll
      for (int i = 0; i < 16; ++i) b[i] = pB[tapOff(kc * 32 + 16 + i)];
    }
    if (kc & 1)
      acc1 = __builtin_amdgcn_wmma_f32_16x16x32_f16(false, a, false, b, (short)0, acc1,
                                                    false, false);
    else
      acc0 = __builtin_amdgcn_wmma_f32_16x16x32_f16(false, a, false, b, (short)0, acc0,
                                                    false, false);
  }

  const float gamma = softplusf(betP[0]) / softplusf(lamP[0]);
  const float invg  = 1.0f / gamma;
  const float invg2 = invg * invg;

  const int oy = y0 + dyp, ot = t0 + dtp;
  const size_t sIdx = (size_t)(ox * PY + oy) * PT + ot;
  const int fbase = lo ? 0 : 8;  // C rows M = i (+8 for hi half-wave)
  const int ixi = ox - 2, iyi = oy - 2, iti = ot - 2;
  const bool inb = (unsigned)ixi < IX && (unsigned)iyi < IY && (unsigned)iti < IT;
  const size_t imIdx = inb ? ((size_t)(ixi * IY + iyi) * IT + iti) : 0;

  float* __restrict__ outP = outB + (size_t)(mbI * NF + fbase) * NPAD + sIdx;
  const float* __restrict__ zP = zBuf + (size_t)(mbI * NF + fbase) * IMS + imIdx;
  const float* __restrict__ uP = uBuf + (size_t)(mbI * NF + fbase) * IMS + imIdx;

#pragma unroll
  for (int i = 0; i < 8; ++i) {
    float cval = acc0[i] + acc1[i];
    size_t o = (size_t)i * NPAD;
    if (mode == 0) {
      float zu = 0.0f;
      if (inb) {
        size_t q = (size_t)i * IMS;
        zu = zP[q] - uP[q];
      }
      outP[o] = cval + gamma * zu;
    } else {
      outP[o] = invg * outP[o] - invg2 * cval;   // in-place Neumann S-step
    }
  }
}

// ---------------------------------------------------------------------------
// 16-channel -> 2-channel circular convolution (sum over each 8-filter half).
//   mode 0: in = b (padded), out t (padded)
//   mode 1: in = z (IM layout, zero outside), out Ds (IM layout)
// ---------------------------------------------------------------------------
__global__ __launch_bounds__(256) void conv16to2_kernel(
    const float* __restrict__ inF,
    const float* __restrict__ dHalf,
    float* __restrict__ outF,
    int mode)
{
  __shared__ float sF[NH * 125];
#if HAVE_TDM
  if (threadIdx.x < 32) tdm_load_f32(sF, dHalf, NH * 125);
#else
  for (int idx = threadIdx.x; idx < NH * 125; idx += 256) sF[idx] = dHalf[idx];
#endif
  __syncthreads();

  const size_t spatial = (mode == 0) ? (size_t)NPAD : (size_t)IMS;
  const size_t total   = (size_t)MB * 2 * spatial;
  size_t gid = (size_t)blockIdx.x * 256 + threadIdx.x;
  if (gid >= total) return;

  int n  = (int)(gid % spatial);
  int hm = (int)(gid / spatial);
  int h = hm & 1, mbI = hm >> 1;

  int oxp, oyp, otp;
  if (mode == 0) {
    oxp = n / (PY * PT); int r = n - oxp * (PY * PT);
    oyp = r / PT; otp = r - oyp * PT;
  } else {
    int ix = n / (IY * IT); int r = n - ix * (IY * IT);
    int iy = r / IT; int it = r - iy * IT;
    oxp = ix + 2; oyp = iy + 2; otp = it + 2;
  }

  // conv source coord per axis: wrap(o + 2 - tap); mode 1 maps into IM (-1 => zero)
  int offX[5], offY[5], offT[5];
#pragma unroll
  for (int d = 0; d < 5; ++d) {
    int vx = wrapi(oxp + 2 - d, PX);
    int vy = wrapi(oyp + 2 - d, PY);
    int vt = wrapi(otp + 2 - d, PT);
    if (mode == 0) {
      offX[d] = vx * (PY * PT);
      offY[d] = vy * PT;
      offT[d] = vt;
    } else {
      int ix = vx - 2, iy = vy - 2, it = vt - 2;
      offX[d] = ((unsigned)ix < IX) ? ix * (IY * IT) : -1;
      offY[d] = ((unsigned)iy < IY) ? iy * IT : -1;
      offT[d] = ((unsigned)it < IT) ? it : -1;
    }
  }

  float acc = 0.0f;
  for (int k = 0; k < NH; ++k) {
    const float* __restrict__ bk = inF + (size_t)(mbI * NF + h * NH + k) * spatial;
    const float* __restrict__ fk = sF + k * 125;
    __builtin_prefetch(bk, 0, 1);
    int fi = 0;
    for (int dx = 0; dx < 5; ++dx) {
      int oX = offX[dx];
      for (int dy = 0; dy < 5; ++dy) {
        int oY = offY[dy];
#pragma unroll
        for (int dt = 0; dt < 5; ++dt, ++fi) {
          int oT = offT[dt];
          if (mode == 0) {
            acc = fmaf(fk[fi], bk[oX + oY + oT], acc);
          } else if ((oX | oY | oT) >= 0) {
            acc = fmaf(fk[fi], bk[oX + oY + oT], acc);
          }
        }
      }
    }
  }
  outF[gid] = acc;
}

// s -> (z,u) soft-threshold + dual update (interior crop of padded s)
__global__ __launch_bounds__(256) void zu_update_kernel(
    const float* __restrict__ sPad,
    float* __restrict__ zBuf,
    float* __restrict__ uBuf,
    const float* __restrict__ alpP,
    const float* __restrict__ betP)
{
  size_t gid = (size_t)blockIdx.x * 256 + threadIdx.x;
  if (gid >= (size_t)MB * NF * IMS) return;
  int n  = (int)(gid % IMS);
  int fm = (int)(gid / IMS);
  int ix = n / (IY * IT); int r = n - ix * (IY * IT);
  int iy = r / IT; int it = r - iy * IT;
  size_t sp = (size_t)fm * NPAD + (size_t)((ix + 2) * PY + (iy + 2)) * PT + (it + 2);
  float thresh = softplusf(alpP[0]) / softplusf(betP[0]);
  float su  = sPad[sp] + uBuf[gid];
  float mag = fabsf(su) - thresh;
  float z   = (mag > 0.0f) ? copysignf(mag, su) : 0.0f;
  zBuf[gid] = z;
  uBuf[gid] = su - z;
}

__global__ __launch_bounds__(256) void x_update_kernel(
    const float* __restrict__ xIn,
    const float* __restrict__ Ds,
    float* __restrict__ xOut,
    const float* __restrict__ lamP)
{
  size_t gid = (size_t)blockIdx.x * 256 + threadIdx.x;
  if (gid >= (size_t)MB * 2 * IMS) return;
  float lam = softplusf(lamP[0]);
  xOut[gid] = (xIn[gid] + lam * Ds[gid]) / (1.0f + lam);
}

__global__ __launch_bounds__(256) void init_state_kernel(
    float* __restrict__ uB, float* __restrict__ zB,
    float* __restrict__ xc, const float* __restrict__ x0)
{
  size_t gid = (size_t)blockIdx.x * 256 + threadIdx.x;
  if (gid < (size_t)MB * NF * IMS) { uB[gid] = 0.0f; zB[gid] = 0.0f; }
  if (gid < (size_t)MB * 2 * IMS)  xc[gid] = x0[gid];
}

extern "C" void kernel_launch(void* const* d_in, const int* in_sizes, int n_in,
                              void* d_out, int out_size, void* d_ws, size_t ws_size,
                              hipStream_t stream) {
  const float* x0   = (const float*)d_in[0];
  const float* dH   = (const float*)d_in[1];
  const float* lamP = (const float*)d_in[2];
  const float* alpP = (const float*)d_in[3];
  const float* betP = (const float*)d_in[4];

  float* ws = (float*)d_ws;
  size_t off = 0;
  float* bB  = ws + off; off += (size_t)MB * NF * NPAD;  // rhs / s (padded)
  float* tB  = ws + off; off += (size_t)MB * 2 * NPAD;   // dHb field (padded)
  float* uB  = ws + off; off += (size_t)MB * NF * IMS;   // dual
  float* zB  = ws + off; off += (size_t)MB * NF * IMS;   // aux
  float* dsB = ws + off; off += (size_t)MB * 2 * IMS;    // Ds (IM)
  float* xc  = ws + off; off += (size_t)MB * 2 * IMS;    // current x

  dim3 blk(256);
  const unsigned initBlocks  = (unsigned)(((size_t)MB * NF * IMS + 255) / 256);
  const unsigned corrBlocks  = TOTAL_TILES / 8;  // 10086, exact
  const unsigned convTBlocks = (unsigned)(((size_t)MB * 2 * NPAD + 255) / 256);
  const unsigned convDBlocks = (unsigned)(((size_t)MB * 2 * IMS + 255) / 256);
  const unsigned zuBlocks    = initBlocks;

  init_state_kernel<<<initBlocks, blk, 0, stream>>>(uB, zB, xc, x0);

  for (int it = 0; it < 2; ++it) {
    // b = corr(x, d) + gamma*(z-u)
    corr16_wmma_kernel<<<corrBlocks, blk, 0, stream>>>(xc, dH, zB, uB, bB, lamP, betP, 0);
    // t_h = sum_k conv(b_k, d_k)
    conv16to2_kernel<<<convTBlocks, blk, 0, stream>>>(bB, dH, tB, 0);
    // s_k = (1/g)*b_k - (1/g^2)*corr(t, d_k)   (Sherman-Morrison, Neumann O(1))
    corr16_wmma_kernel<<<corrBlocks, blk, 0, stream>>>(tB, dH, zB, uB, bB, lamP, betP, 1);
    // z,u update (soft-threshold)
    zu_update_kernel<<<zuBlocks, blk, 0, stream>>>(bB, zB, uB, alpP, betP);
    // Ds_h = sum_k conv(z_k, d_k)  on interior
    conv16to2_kernel<<<convDBlocks, blk, 0, stream>>>(zB, dH, dsB, 1);
    // x = (x + lam*Ds) / (1+lam)
    float* xdst = (it == 1) ? (float*)d_out : xc;
    x_update_kernel<<<convDBlocks, blk, 0, stream>>>(xc, dsB, xdst, lamP);
  }

  (void)in_sizes; (void)n_in; (void)out_size; (void)ws_size; (void)off;
}
